// HFFastMKAAttention_15247133900880
// MI455X (gfx1250) — matrix operand order repaired and checked by reading, not compiled, not run on hardware
//
#include <hip/hip_runtime.h>
#include <hip/hip_bf16.h>

typedef __bf16 bf16_t;
typedef __attribute__((ext_vector_type(16))) __bf16 v16bf;
typedef __attribute__((ext_vector_type(8)))  __bf16 v8bf;
typedef __attribute__((ext_vector_type(8)))  float  v8f;

#define T_SEQ 2048
#define D_MODEL 2048
#define NH 16
#define NKVH 4
#define HD 128

#define BM 128
#define BN 64
#define BK 32

__device__ __forceinline__ float wred_max16(float v) {
  v = fmaxf(v, __shfl_xor(v, 1, 32));
  v = fmaxf(v, __shfl_xor(v, 2, 32));
  v = fmaxf(v, __shfl_xor(v, 4, 32));
  v = fmaxf(v, __shfl_xor(v, 8, 32));
  return v;
}
__device__ __forceinline__ float wred_sum16(float v) {
  v += __shfl_xor(v, 1, 32);
  v += __shfl_xor(v, 2, 32);
  v += __shfl_xor(v, 4, 32);
  v += __shfl_xor(v, 8, 32);
  return v;
}
__device__ __forceinline__ float wred_sum32(float v) {
  v += __shfl_xor(v, 1, 32);
  v += __shfl_xor(v, 2, 32);
  v += __shfl_xor(v, 4, 32);
  v += __shfl_xor(v, 8, 32);
  v += __shfl_xor(v, 16, 32);
  return v;
}

// async 16B global->LDS copy (per lane); tracked by ASYNCcnt
__device__ __forceinline__ void async_cp16(const bf16_t* g, bf16_t* l) {
  unsigned loff = (unsigned)(uintptr_t)l;  // flat LDS pointer: low 32 bits = LDS byte address
  asm volatile("global_load_async_to_lds_b128 %0, %1, off"
               :: "v"(loff), "v"(g)
               : "memory");
}
__device__ __forceinline__ void wait_async0() {
  asm volatile("s_wait_asynccnt 0x0" ::: "memory");
}

// ---------------- elementwise / prep kernels ----------------

// W (K x N, f32 row-major) -> Wt (N x K, bf16 row-major)
__global__ void wtrans_kernel(const float* __restrict__ W, bf16_t* __restrict__ Wt,
                              int K, int N) {
  int i = blockIdx.x * blockDim.x + threadIdx.x;
  if (i >= K * N) return;
  int n = i / K, k = i % K;
  Wt[i] = (bf16_t)W[(size_t)k * N + n];
}

__global__ void f2bf_kernel(const float* __restrict__ x, bf16_t* __restrict__ y, int n) {
  int i = blockIdx.x * blockDim.x + threadIdx.x;
  if (i >= n) return;
  y[i] = (bf16_t)x[i];
}

// causal prefix EMA along T, parallel over D
__global__ void ema_kernel(const float* __restrict__ x, float* __restrict__ l2) {
  int d = blockIdx.x * blockDim.x + threadIdx.x;
  if (d >= D_MODEL) return;
  float m = 0.0f;
  for (int t = 0; t < T_SEQ; ++t) {
    m = 0.9f * m + 0.1f * x[(size_t)t * D_MODEL + d];
    l2[(size_t)t * D_MODEL + d] = m;
  }
}

// lam = softmax(h @ Wr2 + br2); one wave per row
__global__ __launch_bounds__(256) void router2_kernel(
    const bf16_t* __restrict__ Hb, const float* __restrict__ Wr2,
    const float* __restrict__ br2, float* __restrict__ lam) {
  int wid = threadIdx.x >> 5, lane = threadIdx.x & 31;
  int t = blockIdx.x * 8 + wid;
  if (t >= T_SEQ) return;
  float s0 = 0.f, s1 = 0.f;
  for (int j = lane; j < D_MODEL / 2; j += 32) {
    float hv = (float)Hb[(size_t)t * (D_MODEL / 2) + j];
    s0 += hv * Wr2[2 * j];
    s1 += hv * Wr2[2 * j + 1];
  }
  s0 = wred_sum32(s0);
  s1 = wred_sum32(s1);
  if (lane == 0) {
    s0 += br2[0]; s1 += br2[1];
    float mx = fmaxf(s0, s1);
    float e0 = __expf(s0 - mx), e1 = __expf(s1 - mx);
    float inv = 1.0f / (e0 + e1);
    lam[2 * t]     = e0 * inv;
    lam[2 * t + 1] = e1 * inv;
  }
}

__global__ void fuse_kernel(const float* __restrict__ hs, const float* __restrict__ l2,
                            const float* __restrict__ lam, bf16_t* __restrict__ out) {
  int i = blockIdx.x * blockDim.x + threadIdx.x;
  if (i >= T_SEQ * D_MODEL) return;
  int t = i / D_MODEL;
  out[i] = (bf16_t)(lam[2 * t] * hs[i] + lam[2 * t + 1] * l2[i]);
}

// RoPE + head-major relayout: X (T, Hn*HD) bf16 -> Xr (Hn, T, HD) bf16
__global__ void rope_kernel(const bf16_t* __restrict__ X, bf16_t* __restrict__ Xr, int Hn) {
  int i = blockIdx.x * blockDim.x + threadIdx.x;
  int total = Hn * T_SEQ * (HD / 2);
  if (i >= total) return;
  int j = i & 63;
  int rem = i >> 6;
  int t = rem % T_SEQ;
  int h = rem / T_SEQ;
  float inv = __powf(10000.0f, -(float)j * (1.0f / 64.0f));
  float ang = (float)t * inv;
  float c = __cosf(ang), s = __sinf(ang);
  size_t src = (size_t)t * (Hn * HD) + h * HD + j;
  float x1 = (float)X[src];
  float x2 = (float)X[src + 64];
  size_t dst = ((size_t)h * T_SEQ + t) * HD + j;
  Xr[dst]      = (bf16_t)(x1 * c - x2 * s);
  Xr[dst + 64] = (bf16_t)(x2 * c + x1 * s);
}

// Vb (T, KVH*HD) -> Vt (KVH, HD, T)
__global__ void vtrans_kernel(const bf16_t* __restrict__ Vb, bf16_t* __restrict__ Vt) {
  int i = blockIdx.x * blockDim.x + threadIdx.x;
  if (i >= NKVH * HD * T_SEQ) return;
  int t = i % T_SEQ;
  int rest = i / T_SEQ;  // kvh*HD + hd
  Vt[i] = Vb[(size_t)t * (NKVH * HD) + rest];
}

// ---------------- WMMA bf16 GEMM: C(MxN) = A(MxK) @ Bt(NxK)^T ----------------
// Block tile 128x64 staged in LDS via async global->LDS copies (double buffered).
// 8 waves of 32x32 (2x2 WMMA) each. Requires M%128==0, N%64==0, K%32==0.
// mode 0: store bf16; mode 1: silu(x + bias) -> bf16; mode 2: store f32
__global__ __launch_bounds__(256) void gemm_bf16_wmma(
    const bf16_t* __restrict__ A, const bf16_t* __restrict__ Bt,
    int M, int N, int K,
    bf16_t* __restrict__ Cb, float* __restrict__ Cf,
    const float* __restrict__ bias, int mode) {
  __shared__ bf16_t As[2][BM * BK];
  __shared__ bf16_t Bs[2][BN * BK];

  int tid = threadIdx.x;
  int lane = tid & 31;
  int wid = tid >> 5;
  int l16 = lane & 15, half = lane >> 4;
  int wm = wid >> 1, wn = wid & 1;  // 4 x 2 wave grid

  int nbn = N / BN;
  int m0 = (blockIdx.x / nbn) * BM;
  int n0 = (blockIdx.x % nbn) * BN;

  // per-thread 16B async-copy chunks: A = 512 chunks (2/thread), B = 256 (1/thread)
  int arow0 = tid >> 2;
  int arow1 = (tid + 256) >> 2;
  int aseg = (tid & 3) * 8;
  int brow = tid >> 2;
  int bseg = (tid & 3) * 8;

  const bf16_t* Ag = A + (size_t)m0 * K;
  const bf16_t* Bg = Bt + (size_t)n0 * K;

  int nk = K / BK;
  // prologue: k-slice 0 into buffer 0
  async_cp16(Ag + (size_t)arow0 * K + aseg, &As[0][arow0 * BK + aseg]);
  async_cp16(Ag + (size_t)arow1 * K + aseg, &As[0][arow1 * BK + aseg]);
  async_cp16(Bg + (size_t)brow * K + bseg, &Bs[0][brow * BK + bseg]);

  v8f acc[2][2] = {};

  for (int it = 0; it < nk; ++it) {
    int cur = it & 1;
    wait_async0();
    __syncthreads();
    if (it + 1 < nk) {
      int k0 = (it + 1) * BK;
      int nxt = cur ^ 1;
      async_cp16(Ag + (size_t)arow0 * K + k0 + aseg, &As[nxt][arow0 * BK + aseg]);
      async_cp16(Ag + (size_t)arow1 * K + k0 + aseg, &As[nxt][arow1 * BK + aseg]);
      async_cp16(Bg + (size_t)brow * K + k0 + bseg, &Bs[nxt][brow * BK + bseg]);
    }
    // fragments from LDS
    v16bf af[2], bfr[2];
#pragma unroll
    for (int i = 0; i < 2; ++i) {
      const bf16_t* ap = &As[cur][(wm * 32 + i * 16 + l16) * BK + half * 16];
      v8bf lo = *(const v8bf*)ap;
      v8bf hi = *(const v8bf*)(ap + 8);
      af[i] = __builtin_shufflevector(lo, hi, 0, 1, 2, 3, 4, 5, 6, 7,
                                      8, 9, 10, 11, 12, 13, 14, 15);
      const bf16_t* bp = &Bs[cur][(wn * 32 + i * 16 + l16) * BK + half * 16];
      bfr[i] = *(const v16bf*)bp;
    }
#pragma unroll
    for (int i = 0; i < 2; ++i)
#pragma unroll
      for (int j = 0; j < 2; ++j)
        acc[i][j] = __builtin_amdgcn_wmma_f32_16x16x32_bf16(
            false, af[i], false, bfr[j], (short)0, acc[i][j], false, false);
  }

#pragma unroll
  for (int i = 0; i < 2; ++i) {
#pragma unroll
    for (int j = 0; j < 2; ++j) {
#pragma unroll
      for (int r = 0; r < 8; ++r) {
        int row = m0 + wm * 32 + i * 16 + r + half * 8;
        int col = n0 + wn * 32 + j * 16 + l16;
        float v = acc[i][j][r];
        if (mode == 0) {
          Cb[(size_t)row * N + col] = (bf16_t)v;
        } else if (mode == 1) {
          float x = v + bias[col];
          Cb[(size_t)row * N + col] = (bf16_t)(x / (1.0f + __expf(-x)));
        } else {
          Cf[(size_t)row * N + col] = v;
        }
      }
    }
  }
}

// ---------------- flash attention (WMMA, online softmax) ----------------
// Qr (H,T,HD), Kr (KVH,T,HD), Vt (KVH,HD,T) -> Out (T, H*HD), all bf16
__global__ __launch_bounds__(256) void flash_attn_wmma(
    const bf16_t* __restrict__ Qr, const bf16_t* __restrict__ Kr,
    const bf16_t* __restrict__ Vt, bf16_t* __restrict__ Out) {
  __shared__ float Pl[8][16 * 32];
  int wid = threadIdx.x >> 5, lane = threadIdx.x & 31;
  int l16 = lane & 15, half = lane >> 4;
  int w = blockIdx.x * 8 + wid;
  int h = w >> 7;
  int q0 = (w & 127) << 4;
  int kvh = h >> 2;  // H/KVH = 4
  const float scale = 0.08838834764831845f;  // 1/sqrt(128)

  v16bf qf[4];
  const bf16_t* qbase = Qr + ((size_t)h * T_SEQ + q0 + l16) * HD + half * 16;
#pragma unroll
  for (int c = 0; c < 4; ++c) {
    v8bf lo = *(const v8bf*)(qbase + c * 32);
    v8bf hi = *(const v8bf*)(qbase + c * 32 + 8);
    qf[c] = __builtin_shufflevector(lo, hi, 0, 1, 2, 3, 4, 5, 6, 7,
                                    8, 9, 10, 11, 12, 13, 14, 15);
  }

  v8f O[8] = {};
  float rowmax[8], rowsum[8];
#pragma unroll
  for (int r = 0; r < 8; ++r) { rowmax[r] = -1e30f; rowsum[r] = 0.0f; }

  const bf16_t* kbase = Kr + (size_t)kvh * T_SEQ * HD;
  const bf16_t* vbase = Vt + (size_t)kvh * HD * T_SEQ;
  float* pl = Pl[wid];

  for (int kb = 0; kb < q0 + 16; kb += 32) {
    v8f S0 = {}, S1 = {};
#pragma unroll
    for (int c = 0; c < 4; ++c) {
      v16bf b0 = *(const v16bf*)(kbase + (size_t)(kb + l16) * HD + c * 32 + half * 16);
      v16bf b1 = *(const v16bf*)(kbase + (size_t)(kb + 16 + l16) * HD + c * 32 + half * 16);
      S0 = __builtin_amdgcn_wmma_f32_16x16x32_bf16(false, qf[c], false, b0,
                                                   (short)0, S0, false, false);
      S1 = __builtin_amdgcn_wmma_f32_16x16x32_bf16(false, qf[c], false, b1,
                                                   (short)0, S1, false, false);
    }
    float corr[8];
#pragma unroll
    for (int r = 0; r < 8; ++r) {
      int mrow = q0 + r + half * 8;
      float a0 = S0[r] * scale;
      float a1 = S1[r] * scale;
      if (kb + l16 > mrow) a0 = -1e30f;
      if (kb + 16 + l16 > mrow) a1 = -1e30f;
      float mx = wred_max16(fmaxf(a0, a1));
      float mnew = fmaxf(rowmax[r], mx);
      float c0 = __expf(rowmax[r] - mnew);
      float e0 = __expf(a0 - mnew);
      float e1 = __expf(a1 - mnew);
      float sr = wred_sum16(e0 + e1);
      rowsum[r] = rowsum[r] * c0 + sr;
      rowmax[r] = mnew;
      corr[r] = c0;
      int rl = r + half * 8;
      pl[rl * 32 + l16] = e0;
      pl[rl * 32 + 16 + l16] = e1;
    }
#pragma unroll
    for (int j = 0; j < 8; ++j) {
#pragma unroll
      for (int r = 0; r < 8; ++r) O[j][r] *= corr[r];
    }
    v16bf pf;
#pragma unroll
    for (int i = 0; i < 16; ++i) {
      int k = half * 16 + i;
      pf[i] = (bf16_t)pl[l16 * 32 + k];
    }
#pragma unroll
    for (int j = 0; j < 8; ++j) {
      v16bf bv = *(const v16bf*)(vbase + (size_t)(j * 16 + l16) * T_SEQ + kb + half * 16);
      O[j] = __builtin_amdgcn_wmma_f32_16x16x32_bf16(false, pf, false, bv,
                                                     (short)0, O[j], false, false);
    }
  }
#pragma unroll
  for (int j = 0; j < 8; ++j) {
#pragma unroll
    for (int r = 0; r < 8; ++r) {
      int t = q0 + r + half * 8;
      int col = h * HD + j * 16 + l16;
      Out[(size_t)t * (NH * HD) + col] = (bf16_t)(O[j][r] / rowsum[r]);
    }
  }
}

// ---------------- host side ----------------

extern "C" void kernel_launch(void* const* d_in, const int* in_sizes, int n_in,
                              void* d_out, int out_size, void* d_ws, size_t ws_size,
                              hipStream_t stream) {
  (void)in_sizes; (void)n_in; (void)out_size; (void)ws_size;
  const float* hs  = (const float*)d_in[0];
  const float* Wq  = (const float*)d_in[1];
  const float* Wk  = (const float*)d_in[2];
  const float* Wv  = (const float*)d_in[3];
  const float* Wo  = (const float*)d_in[4];
  const float* Wr1 = (const float*)d_in[5];
  const float* br1 = (const float*)d_in[6];
  const float* Wr2 = (const float*)d_in[7];
  const float* br2 = (const float*)d_in[8];
  float* out = (float*)d_out;

  char* p = (char*)d_ws;
  auto alloc = [&](size_t bytes) -> void* {
    void* r = (void*)p;
    p += (bytes + 255) & ~(size_t)255;
    return r;
  };
  const size_t TD = (size_t)T_SEQ * D_MODEL;
  float*  l2    = (float*)alloc(TD * 4);
  bf16_t* hidb  = (bf16_t*)alloc(TD * 2);
  bf16_t* Wqt   = (bf16_t*)alloc((size_t)D_MODEL * (NH * HD) * 2);
  bf16_t* Wkt   = (bf16_t*)alloc((size_t)D_MODEL * (NKVH * HD) * 2);
  bf16_t* Wvt   = (bf16_t*)alloc((size_t)D_MODEL * (NKVH * HD) * 2);
  bf16_t* Wr1t  = (bf16_t*)alloc((size_t)D_MODEL * (D_MODEL / 2) * 2);
  bf16_t* Wot   = (bf16_t*)alloc((size_t)(NH * HD) * D_MODEL * 2);
  bf16_t* Qb    = (bf16_t*)alloc((size_t)T_SEQ * (NH * HD) * 2);
  bf16_t* Hb    = (bf16_t*)alloc((size_t)T_SEQ * (D_MODEL / 2) * 2);
  float*  lam   = (float*)alloc((size_t)T_SEQ * 2 * 4);
  bf16_t* fused = (bf16_t*)alloc(TD * 2);
  bf16_t* Kb    = (bf16_t*)alloc((size_t)T_SEQ * (NKVH * HD) * 2);
  bf16_t* Vb    = (bf16_t*)alloc((size_t)T_SEQ * (NKVH * HD) * 2);
  bf16_t* Qrp   = (bf16_t*)alloc((size_t)NH * T_SEQ * HD * 2);
  bf16_t* Krp   = (bf16_t*)alloc((size_t)NKVH * T_SEQ * HD * 2);
  bf16_t* Vtp   = (bf16_t*)alloc((size_t)NKVH * HD * T_SEQ * 2);
  bf16_t* AOut  = (bf16_t*)alloc((size_t)T_SEQ * (NH * HD) * 2);

  dim3 blk(256);
  auto nb = [](size_t n) { return dim3((unsigned)((n + 255) / 256)); };
  auto gg = [](int M, int N) { return dim3((unsigned)((M / BM) * (N / BN))); };

  // weight transposes (f32 -> bf16, NxK)
  wtrans_kernel<<<nb((size_t)D_MODEL * (NH * HD)), blk, 0, stream>>>(Wq, Wqt, D_MODEL, NH * HD);
  wtrans_kernel<<<nb((size_t)D_MODEL * (NKVH * HD)), blk, 0, stream>>>(Wk, Wkt, D_MODEL, NKVH * HD);
  wtrans_kernel<<<nb((size_t)D_MODEL * (NKVH * HD)), blk, 0, stream>>>(Wv, Wvt, D_MODEL, NKVH * HD);
  wtrans_kernel<<<nb((size_t)D_MODEL * (D_MODEL / 2)), blk, 0, stream>>>(Wr1, Wr1t, D_MODEL, D_MODEL / 2);
  wtrans_kernel<<<nb((size_t)(NH * HD) * D_MODEL), blk, 0, stream>>>(Wo, Wot, NH * HD, D_MODEL);

  // activations
  f2bf_kernel<<<nb(TD), blk, 0, stream>>>(hs, hidb, (int)TD);
  ema_kernel<<<nb(D_MODEL), blk, 0, stream>>>(hs, l2);

  // Q = hidden @ Wq
  gemm_bf16_wmma<<<gg(T_SEQ, NH * HD), blk, 0, stream>>>(
      hidb, Wqt, T_SEQ, NH * HD, D_MODEL, Qb, nullptr, nullptr, 0);

  // h = silu(Q @ Wr1 + br1)
  gemm_bf16_wmma<<<gg(T_SEQ, D_MODEL / 2), blk, 0, stream>>>(
      Qb, Wr1t, T_SEQ, D_MODEL / 2, D_MODEL, Hb, nullptr, br1, 1);

  // lam = softmax(h @ Wr2 + br2)
  router2_kernel<<<dim3(T_SEQ / 8), blk, 0, stream>>>(Hb, Wr2, br2, lam);

  // fused = lam0*hidden + lam1*l2
  fuse_kernel<<<nb(TD), blk, 0, stream>>>(hs, l2, lam, fused);

  // K, V = fused @ Wk / Wv
  gemm_bf16_wmma<<<gg(T_SEQ, NKVH * HD), blk, 0, stream>>>(
      fused, Wkt, T_SEQ, NKVH * HD, D_MODEL, Kb, nullptr, nullptr, 0);
  gemm_bf16_wmma<<<gg(T_SEQ, NKVH * HD), blk, 0, stream>>>(
      fused, Wvt, T_SEQ, NKVH * HD, D_MODEL, Vb, nullptr, nullptr, 0);

  // RoPE + relayout
  rope_kernel<<<nb((size_t)NH * T_SEQ * (HD / 2)), blk, 0, stream>>>(Qb, Qrp, NH);
  rope_kernel<<<nb((size_t)NKVH * T_SEQ * (HD / 2)), blk, 0, stream>>>(Kb, Krp, NKVH);
  vtrans_kernel<<<nb((size_t)NKVH * HD * T_SEQ), blk, 0, stream>>>(Vb, Vtp);

  // causal flash attention
  flash_attn_wmma<<<dim3((NH * (T_SEQ / 16)) / 8), blk, 0, stream>>>(Qrp, Krp, Vtp, AOut);

  // out = attn_out @ Wo (f32)
  gemm_bf16_wmma<<<gg(T_SEQ, D_MODEL), blk, 0, stream>>>(
      AOut, Wot, T_SEQ, D_MODEL, NH * HD, nullptr, out, nullptr, 2);
}
